// TransitionDown_75213467287645
// MI455X (gfx1250) — matrix operand order repaired and checked by reading, not compiled, not run on hardware
//
#include <hip/hip_runtime.h>
#include <hip/hip_bf16.h>

typedef __attribute__((ext_vector_type(16))) _Float16 v16h;
typedef __attribute__((ext_vector_type(8)))  float    v8f;

#define G1 256              // blocks for the two GEMM passes
#define WPB 8               // waves per block (256 threads, wave32)
#define TOTAL_WAVES (G1 * WPB)
#define KPAD 96             // 67 features -> pad to 3 x K32 tiles (zero col at k=3)
#define COUT 128
#define NS 16

union ABu { v16h v; uint4 u[2]; };

__device__ __forceinline__ unsigned short f2us(float f) {
    union { _Float16 h; unsigned short u; } t; t.h = (_Float16)f; return t.u;
}
__device__ __forceinline__ unsigned int pack2(float a, float b) {
    union { _Float16 h[2]; unsigned int u; } t;
    t.h[0] = (_Float16)a; t.h[1] = (_Float16)b; return t.u;
}
// exchange with lane^16 via ds_swizzle SWAPX16 (xor=0x10, and=0x1f)
__device__ __forceinline__ float xhalf(float v) {
    return __int_as_float(__builtin_amdgcn_ds_swizzle(__float_as_int(v), 0x401F));
}

// Build padded f16 weight matrix Wh[128][96]: k<3 -> W[o][k], k==3 -> 0,
// 4<=k<68 -> W[o][k-1], else 0.  (zero column keeps LDS x-stores 8B aligned)
__global__ void td_prep(const float* __restrict__ W, unsigned short* __restrict__ Wh) {
    int o = threadIdx.x;            // 128 threads
    for (int k = 0; k < KPAD; ++k) {
        float v = 0.f;
        if (k < 3)                v = W[o * 67 + k];
        else if (k >= 4 && k < 68) v = W[o * 67 + (k - 1)];
        Wh[o * KPAD + k] = f2us(v);
    }
}

// Per-channel scale/shift from deterministic block partials.
__global__ void td_stats(const float* __restrict__ partial,
                         const float* __restrict__ gamma,
                         const float* __restrict__ beta,
                         float* __restrict__ scale, float* __restrict__ shift,
                         int nblocks, float invCnt) {
    int o = threadIdx.x;            // 128 threads
    float S = 0.f, Q = 0.f;
    for (int b = 0; b < nblocks; ++b) {       // fixed order -> deterministic
        S += partial[b * 256 + o];
        Q += partial[b * 256 + 128 + o];
    }
    float mean = S * invCnt;
    float var  = Q * invCnt - mean * mean;
    float inv  = rsqrtf(var + 1e-5f);
    float scl  = gamma[o] * inv;
    scale[o] = scl;
    shift[o] = beta[o] - mean * scl;
}

// PASS1: accumulate per-channel sum / sumsq.  PASS2: affine+ReLU+rowmax -> out.
template <bool PASS1>
__global__ __launch_bounds__(256) void td_gemm(
        const float* __restrict__ p, const float* __restrict__ x,
        const float* __restrict__ np, const int* __restrict__ idx,
        const unsigned short* __restrict__ Wh,
        const float* __restrict__ scale, const float* __restrict__ shift,
        float* __restrict__ partial, float* __restrict__ out, int M) {
    __shared__ unsigned short sfeat[WPB * NS * KPAD];   // 24576 B
    __shared__ float psum[WPB][COUT];                   // pass1 block reduce
    __shared__ float psq [WPB][COUT];

    const int tid  = threadIdx.x;
    const int lane = tid & 31;
    const int w    = tid >> 5;
    const int r    = lane & 15;     // row (neighbor) / column within N-tile
    const int h    = lane >> 4;     // half-wave selector
    const int gwave = blockIdx.x * WPB + w;

    // zero the feature staging region once (pad columns stay zero forever)
    for (int i = tid; i < WPB * NS * KPAD / 2; i += 256)
        ((unsigned int*)sfeat)[i] = 0u;
    __syncthreads();

    unsigned short* lds = sfeat + w * NS * KPAD;

    // B fragments, register resident: lane holds column o = n*16+r,
    // lanes 0-15 K {0-7,16-23}, lanes 16-31 K {8-15,24-31} (matches A)
    v16h bfr[3][8];
#pragma unroll
    for (int n = 0; n < 8; ++n) {
        const unsigned short* Wrow = Wh + (n * 16 + r) * KPAD;
#pragma unroll
        for (int t = 0; t < 3; ++t) {
            ABu tmp;
            tmp.u[0] = *(const uint4*)(Wrow + t * 32 + h * 8);
            tmp.u[1] = *(const uint4*)(Wrow + t * 32 + 16 + h * 8);
            bfr[t][n] = tmp.v;
        }
    }

    float sc[8], sh[8];
    float sumA[8], sqA[8];
#pragma unroll
    for (int n = 0; n < 8; ++n) {
        sumA[n] = 0.f; sqA[n] = 0.f;
        if (!PASS1) { sc[n] = scale[n * 16 + r]; sh[n] = shift[n * 16 + r]; }
    }

    for (int m = gwave; m < M; m += TOTAL_WAVES) {
        const int gi = idx[m * NS + r];
        // stage rel = p[gi] - n_p[m] at k=0..2 (lanes 0-15 own a row each)
        if (lane < 16) {
#pragma unroll
            for (int d = 0; d < 3; ++d)
                lds[r * KPAD + d] = f2us(p[(size_t)gi * 3 + d] - np[(size_t)m * 3 + d]);
        }
        // stage x[gi][0..63] at k=4..67; lane handles 32 floats (its K-half)
        const float4* xp = (const float4*)(x + (size_t)gi * 64 + h * 32);
#pragma unroll
        for (int j = 0; j < 8; ++j) {
            float4 v = xp[j];
            uint2 s; s.x = pack2(v.x, v.y); s.y = pack2(v.z, v.w);
            *(uint2*)(lds + r * KPAD + 4 + h * 32 + j * 4) = s;
        }
        // wave-synchronous cross-lane LDS handoff
        asm volatile("s_wait_dscnt 0" ::: "memory");

        // A fragments (ISA 16-bit A layout): lanes 0-15 K {0-7,16-23},
        // lanes 16-31 K {8-15,24-31} of each K32 tile
        v16h afr[3];
        const unsigned short* rowp = lds + r * KPAD;
#pragma unroll
        for (int t = 0; t < 3; ++t) {
            ABu tmp;
            tmp.u[0] = *(const uint4*)(rowp + t * 32 + h * 8);
            tmp.u[1] = *(const uint4*)(rowp + t * 32 + 16 + h * 8);
            afr[t] = tmp.v;
        }

#pragma unroll
        for (int n = 0; n < 8; ++n) {
            v8f acc = {};
#pragma unroll
            for (int t = 0; t < 3; ++t)
                acc = __builtin_amdgcn_wmma_f32_16x16x32_f16(
                        false, afr[t], false, bfr[t][n], (short)0, acc, false, false);
            if (PASS1) {
                float s = 0.f, q = 0.f;
#pragma unroll
                for (int j = 0; j < 8; ++j) { s += acc[j]; q += acc[j] * acc[j]; }
                sumA[n] += s; sqA[n] += q;
            } else {
                float mx = 0.f;   // == relu floor, since max over 16 relu'd values
#pragma unroll
                for (int j = 0; j < 8; ++j)
                    mx = fmaxf(mx, acc[j] * sc[n] + sh[n]);
                mx = fmaxf(mx, xhalf(mx));     // fold rows 8-15 half-lane
                if (lane < 16) out[(size_t)m * COUT + n * 16 + lane] = mx;
            }
        }
    }

    if (PASS1) {
#pragma unroll
        for (int n = 0; n < 8; ++n) {
            float s = sumA[n] + xhalf(sumA[n]);
            float q = sqA[n]  + xhalf(sqA[n]);
            if (lane < 16) { psum[w][n * 16 + lane] = s; psq[w][n * 16 + lane] = q; }
        }
        __syncthreads();
        if (tid < COUT) {                       // fixed-order block reduction
            float S = 0.f, Q = 0.f;
            for (int ww = 0; ww < WPB; ++ww) { S += psum[ww][tid]; Q += psq[ww][tid]; }
            partial[blockIdx.x * 256 + tid]       = S;
            partial[blockIdx.x * 256 + 128 + tid] = Q;
        }
    }
}

extern "C" void kernel_launch(void* const* d_in, const int* in_sizes, int n_in,
                              void* d_out, int out_size, void* d_ws, size_t ws_size,
                              hipStream_t stream) {
    const float* p     = (const float*)d_in[0];
    const float* x     = (const float*)d_in[1];
    const float* np    = (const float*)d_in[2];
    const int*   idx   = (const int*)d_in[3];
    const float* W     = (const float*)d_in[4];
    const float* gamma = (const float*)d_in[5];
    const float* beta  = (const float*)d_in[6];
    float* out = (float*)d_out;
    const int M = in_sizes[2] / 3;     // n_p is [M,3]

    // ws layout: Wh f16[128*96] | scale f32[128] | shift f32[128] | partials f32[G1*256]
    unsigned short* Wh = (unsigned short*)d_ws;
    float* scale   = (float*)((char*)d_ws + COUT * KPAD * 2);
    float* shift   = scale + COUT;
    float* partial = shift + COUT;

    td_prep<<<1, 128, 0, stream>>>(W, Wh);
    td_gemm<true><<<G1, 256, 0, stream>>>(p, x, np, idx, Wh, nullptr, nullptr, partial, nullptr, M);
    td_stats<<<1, 128, 0, stream>>>(partial, gamma, beta, scale, shift, G1, 1.f / (16.f * (float)M));
    td_gemm<false><<<G1, 256, 0, stream>>>(p, x, np, idx, Wh, scale, shift, nullptr, out, M);
}
